// AgentAttention_59468117180775
// MI455X (gfx1250) — matrix operand order repaired and checked by reading, not compiled, not run on hardware
//
#include <hip/hip_runtime.h>

#define BATCH   16
#define SEQ     1024
#define DMODEL  256
#define NHEADS  8
#define HDIM    32
#define WSZ     (DMODEL*DMODEL)                // 65536
#define PPS     (4*WSZ + 4*DMODEL)
#define QSCALE  (0.17677669529663688f * 1.4426950408889634f)   // (1/sqrt(32)) * log2(e)

typedef __bf16 bf16_t;
typedef __attribute__((ext_vector_type(16))) __bf16 v16bf;
typedef __attribute__((ext_vector_type(8)))  __bf16 v8bf;
typedef __attribute__((ext_vector_type(8)))  float  v8f;

__device__ inline v8f wmma_bf16(v16bf a, v16bf b, v8f c) {
  return __builtin_amdgcn_wmma_f32_16x16x32_bf16(false, a, false, b, (short)0, c, false, false);
}

// A-fragment (16x32, M x K) from a bf16 row-major matrix (rows are M, K contiguous).
__device__ inline v16bf load_a_bf16(const bf16_t* __restrict__ base, int ld, int row0, int k0, int lane) {
  int m = lane & 15, hh = lane >> 4;
  const bf16_t* p = base + (size_t)(row0 + m) * ld + k0 + hh * 16;
  v8bf lo = *(const v8bf*)p;
  v8bf hi = *(const v8bf*)(p + 8);
  return __builtin_shufflevector(lo, hi, 0,1,2,3,4,5,6,7,8,9,10,11,12,13,14,15);
}

// B-fragment (32x16, K x N): B[k][n] = src[(n0+n)*ld + k0 + k]  (src rows are N, K contiguous).
__device__ inline v16bf load_b_rows_bf16(const bf16_t* __restrict__ base, int ld, int n0, int k0, int lane) {
  const bf16_t* p = base + (size_t)(n0 + (lane & 15)) * ld + k0 + (lane >> 4) * 16;
  return *(const v16bf*)p;
}

// ---------------- Kernel 0: one-shot bf16 conversion ----------------------
// Converts x -> xbf and the four weight matrices per batch -> Wbf
// (layout per batch: [Wq|Wk|Wv|Wo], Wq pre-scaled by QSCALE).
__global__ __launch_bounds__(256) void convert_kernel(const float* __restrict__ x,
                                                      const float* __restrict__ fp,
                                                      bf16_t* __restrict__ xbf,
                                                      bf16_t* __restrict__ Wbf) {
  const size_t i = ((size_t)blockIdx.x * 256 + threadIdx.x) * 8;
  {
    v8f v = *(const v8f*)(x + i);
    v8bf o;
#pragma unroll
    for (int j = 0; j < 8; ++j) o[j] = (__bf16)v[j];
    *(v8bf*)(xbf + i) = o;
  }
  {
    const int b = (int)(i >> 18);            // / (4*WSZ)
    const int w = (int)(i & (4 * WSZ - 1));
    const float s = (w < WSZ) ? QSCALE : 1.0f;
    v8f v = *(const v8f*)(fp + (size_t)b * PPS + w);
    v8bf o;
#pragma unroll
    for (int j = 0; j < 8; ++j) o[j] = (__bf16)(v[j] * s);
    *(v8bf*)(Wbf + i) = o;
  }
}

// ---------------- Kernel 1: Q,K,V projections -----------------------------
// grid = BATCH * SEQ/16 blocks, 256 threads (8 waves); wave w owns n-tiles {2w, 2w+1}.
// V is stored transposed: Vt[b][h][d][s].
__global__ __launch_bounds__(256) void qkv_kernel(const bf16_t* __restrict__ xbf,
                                                  const bf16_t* __restrict__ Wbf,
                                                  const float* __restrict__ fp,
                                                  bf16_t* __restrict__ Qb,
                                                  bf16_t* __restrict__ Kb,
                                                  bf16_t* __restrict__ Vt) {
  __shared__ float lds[8][16][17];
  const int wave = threadIdx.x >> 5, lane = threadIdx.x & 31;
  const int bidx = blockIdx.x >> 6;
  const int s0   = (blockIdx.x & 63) * 16;
  const bf16_t* xb  = xbf + (size_t)bidx * SEQ * DMODEL;
  const bf16_t* Wqb = Wbf + (size_t)bidx * 4 * WSZ;
  const bf16_t* Wkb = Wqb + WSZ;
  const bf16_t* Wvb = Wqb + 2 * WSZ;
  const float*  bq  = fp + (size_t)bidx * PPS + 4 * WSZ;
  const float*  bk  = bq + DMODEL;
  const float*  bv  = bk + DMODEL;

  v8f accQ[2] = {}, accK[2] = {}, accV[2] = {};
  const int nt0 = wave * 2;

  for (int kk = 0; kk < DMODEL; kk += 32) {
    v16bf a = load_a_bf16(xb, DMODEL, s0, kk, lane);
#pragma unroll
    for (int t = 0; t < 2; ++t) {
      const int n0 = (nt0 + t) * 16;
      accQ[t] = wmma_bf16(a, load_b_rows_bf16(Wqb, DMODEL, n0, kk, lane), accQ[t]);
      accK[t] = wmma_bf16(a, load_b_rows_bf16(Wkb, DMODEL, n0, kk, lane), accK[t]);
      accV[t] = wmma_bf16(a, load_b_rows_bf16(Wvb, DMODEL, n0, kk, lane), accV[t]);
    }
  }

  const int n = lane & 15, hh = lane >> 4;

#pragma unroll
  for (int t = 0; t < 2; ++t) {
    const int n0 = (nt0 + t) * 16;
    { // Q: scaled bias, LDS transpose, coalesced store
      const float bias = bq[n0 + n] * QSCALE;
      __syncthreads();
#pragma unroll
      for (int r = 0; r < 8; ++r) lds[wave][hh * 8 + r][n] = accQ[t][r] + bias;
      __syncthreads();
      v8bf o;
#pragma unroll
      for (int i = 0; i < 8; ++i) o[i] = (__bf16)lds[wave][n][hh * 8 + i];
      *(v8bf*)(Qb + ((size_t)bidx * SEQ + s0 + n) * DMODEL + n0 + hh * 8) = o;
    }
    { // K
      const float bias = bk[n0 + n];
      __syncthreads();
#pragma unroll
      for (int r = 0; r < 8; ++r) lds[wave][hh * 8 + r][n] = accK[t][r] + bias;
      __syncthreads();
      v8bf o;
#pragma unroll
      for (int i = 0; i < 8; ++i) o[i] = (__bf16)lds[wave][n][hh * 8 + i];
      *(v8bf*)(Kb + ((size_t)bidx * SEQ + s0 + n) * DMODEL + n0 + hh * 8) = o;
    }
    { // Vt: lane holds a fixed column e -> fixed (head, d), 8 consecutive s values.
      const int e = n0 + n;
      const int head = e >> 5, d = e & 31;
      const float bias = bv[e];
      v8bf o;
#pragma unroll
      for (int r = 0; r < 8; ++r) o[r] = (__bf16)(accV[t][r] + bias);
      *(v8bf*)(Vt + (((size_t)bidx * NHEADS + head) * HDIM + d) * SEQ + s0 + hh * 8) = o;
    }
  }
}

// ---------------- Kernel 2: flash attention, 64-key chunks ----------------
// One wave per (b, h, 16-query tile). Scores computed transposed (S^T = K*Q^T)
// so the exp'ed probabilities land directly in the A-fragment layout of P@V.
__global__ __launch_bounds__(256) void attn_kernel(const bf16_t* __restrict__ Qb,
                                                   const bf16_t* __restrict__ Kb,
                                                   const bf16_t* __restrict__ Vt,
                                                   bf16_t* __restrict__ Ctx) {
  __shared__ float lds[8][16][33];
  const int wave = threadIdx.x >> 5, lane = threadIdx.x & 31;
  const int tile = blockIdx.x * 8 + wave;
  const int bidx = tile >> 9;           // / (NHEADS * SEQ/16)
  const int rem  = tile & 511;
  const int h    = rem >> 6;
  const int q0   = (rem & 63) * 16;
  const int n = lane & 15, hh = lane >> 4;

  const bf16_t* Qp = Qb + (size_t)bidx * SEQ * DMODEL + h * HDIM;
  const bf16_t* Kp = Kb + (size_t)bidx * SEQ * DMODEL + h * HDIM;
  const bf16_t* Vp = Vt + ((size_t)bidx * NHEADS + h) * HDIM * SEQ;

  // Q as B-fragment of S^T: lane n holds query q0+n, hd-contiguous.
  const v16bf qf = *(const v16bf*)(Qp + (size_t)(q0 + n) * DMODEL + hh * 16);

  float m = -__builtin_inff(), l = 0.f;
  v8f ctx0 = {}, ctx1 = {};

  for (int kb = 0; kb < SEQ; kb += 64) {
    // 4 score WMMAs: keys kb .. kb+63
    v16bf a0 = load_a_bf16(Kp, DMODEL, kb,      0, lane);
    v16bf a1 = load_a_bf16(Kp, DMODEL, kb + 16, 0, lane);
    v16bf a2 = load_a_bf16(Kp, DMODEL, kb + 32, 0, lane);
    v16bf a3 = load_a_bf16(Kp, DMODEL, kb + 48, 0, lane);
    v8f s0v = wmma_bf16(a0, qf, (v8f){});
    v8f s1v = wmma_bf16(a1, qf, (v8f){});
    v8f s2v = wmma_bf16(a2, qf, (v8f){});
    v8f s3v = wmma_bf16(a3, qf, (v8f){});

    // prefetch next chunk of K / V streams (global_prefetch_b8)
    if (kb + 64 < SEQ) {
      __builtin_prefetch(Kp + (size_t)(kb + 64 + n) * DMODEL, 0, 3);
      __builtin_prefetch(Vp + (size_t)n * SEQ + kb + 64, 0, 3);
    }

    // Per-lane: one query (lane&15), 32 of its 64 scores; partner lane has the rest.
    float mx = s0v[0];
#pragma unroll
    for (int r = 0; r < 8; ++r) {
      mx = fmaxf(mx, s0v[r]); mx = fmaxf(mx, s1v[r]);
      mx = fmaxf(mx, s2v[r]); mx = fmaxf(mx, s3v[r]);
    }
    mx = fmaxf(mx, __shfl_xor(mx, 16, 32));
    const float mnew = fmaxf(m, mx);
    const float c = exp2f(m - mnew);
    float p0[8], p1[8], p2[8], p3[8], sum = 0.f;
#pragma unroll
    for (int r = 0; r < 8; ++r) {
      p0[r] = exp2f(s0v[r] - mnew);
      p1[r] = exp2f(s1v[r] - mnew);
      p2[r] = exp2f(s2v[r] - mnew);
      p3[r] = exp2f(s3v[r] - mnew);
      sum += (p0[r] + p1[r]) + (p2[r] + p3[r]);
    }
    sum += __shfl_xor(sum, 16, 32);
    l = l * c + sum;
    m = mnew;

    // P A-fragments: C-layout of S^T matches A-layout of P exactly.
    v16bf pf0, pf1;
#pragma unroll
    for (int r = 0; r < 8; ++r) {
      pf0[r] = (__bf16)p0[r]; pf0[r + 8] = (__bf16)p1[r];
      pf1[r] = (__bf16)p2[r]; pf1[r + 8] = (__bf16)p3[r];
    }

    // Rescale accumulators once per 64 keys: row r of ctx C-layout is query hh*8+r.
#pragma unroll
    for (int r = 0; r < 8; ++r) {
      const float cr = __shfl(c, hh * 8 + r, 32);
      ctx0[r] *= cr; ctx1[r] *= cr;
    }
    v16bf b0 = __builtin_shufflevector(*(const v8bf*)(Vp + (size_t)n * SEQ        + kb      + hh * 8), *(const v8bf*)(Vp + (size_t)n * SEQ        + kb + 16 + hh * 8), 0,1,2,3,4,5,6,7,8,9,10,11,12,13,14,15);
    v16bf b1 = __builtin_shufflevector(*(const v8bf*)(Vp + (size_t)(16 + n) * SEQ + kb      + hh * 8), *(const v8bf*)(Vp + (size_t)(16 + n) * SEQ + kb + 16 + hh * 8), 0,1,2,3,4,5,6,7,8,9,10,11,12,13,14,15);
    v16bf b2 = __builtin_shufflevector(*(const v8bf*)(Vp + (size_t)n * SEQ        + kb + 32 + hh * 8), *(const v8bf*)(Vp + (size_t)n * SEQ        + kb + 48 + hh * 8), 0,1,2,3,4,5,6,7,8,9,10,11,12,13,14,15);
    v16bf b3 = __builtin_shufflevector(*(const v8bf*)(Vp + (size_t)(16 + n) * SEQ + kb + 32 + hh * 8), *(const v8bf*)(Vp + (size_t)(16 + n) * SEQ + kb + 48 + hh * 8), 0,1,2,3,4,5,6,7,8,9,10,11,12,13,14,15);
    ctx0 = wmma_bf16(pf0, b0, ctx0);
    ctx1 = wmma_bf16(pf0, b1, ctx1);
    ctx0 = wmma_bf16(pf1, b2, ctx0);
    ctx1 = wmma_bf16(pf1, b3, ctx1);
  }

  const float linv = 1.f / l;
#pragma unroll
  for (int r = 0; r < 8; ++r) {
    const float lr = __shfl(linv, hh * 8 + r, 32);
    ctx0[r] *= lr; ctx1[r] *= lr;
  }

  // Transpose both hd n-tiles through LDS, store coalesced bf16 rows.
#pragma unroll
  for (int r = 0; r < 8; ++r) {
    lds[wave][hh * 8 + r][n]      = ctx0[r];
    lds[wave][hh * 8 + r][16 + n] = ctx1[r];
  }
  __syncthreads();
  v16bf o;
#pragma unroll
  for (int i = 0; i < 16; ++i) o[i] = (__bf16)lds[wave][n][hh * 16 + i];
  *(v16bf*)(Ctx + ((size_t)bidx * SEQ + q0 + n) * DMODEL + h * HDIM + hh * 16) = o;
}

// ---------------- Kernel 3: output projection -----------------------------
__global__ __launch_bounds__(256) void proj_kernel(const bf16_t* __restrict__ Ctx,
                                                   const bf16_t* __restrict__ Wbf,
                                                   const float* __restrict__ fp,
                                                   float* __restrict__ out) {
  __shared__ float lds[8][16][17];
  const int wave = threadIdx.x >> 5, lane = threadIdx.x & 31;
  const int bidx = blockIdx.x >> 6;
  const int s0   = (blockIdx.x & 63) * 16;
  const bf16_t* Wob = Wbf + (size_t)bidx * 4 * WSZ + 3 * WSZ;
  const float*  bo  = fp + (size_t)bidx * PPS + 4 * WSZ + 3 * DMODEL;
  const bf16_t* C   = Ctx + (size_t)bidx * SEQ * DMODEL;

  v8f acc[2] = {};
  const int nt0 = wave * 2;
  for (int kk = 0; kk < DMODEL; kk += 32) {
    v16bf a = load_a_bf16(C, DMODEL, s0, kk, lane);
#pragma unroll
    for (int t = 0; t < 2; ++t)
      acc[t] = wmma_bf16(a, load_b_rows_bf16(Wob, DMODEL, (nt0 + t) * 16, kk, lane), acc[t]);
  }

  const int n = lane & 15, hh = lane >> 4;
#pragma unroll
  for (int t = 0; t < 2; ++t) {
    const int n0 = (nt0 + t) * 16;
    const float bias = bo[n0 + n];
    __syncthreads();
#pragma unroll
    for (int r = 0; r < 8; ++r) lds[wave][hh * 8 + r][n] = acc[t][r] + bias;
    __syncthreads();
    v8f o;
#pragma unroll
    for (int i = 0; i < 8; ++i) o[i] = lds[wave][n][hh * 8 + i];
    *(v8f*)(out + ((size_t)bidx * SEQ + s0 + n) * DMODEL + n0 + hh * 8) = o;
  }
}

extern "C" void kernel_launch(void* const* d_in, const int* in_sizes, int n_in,
                              void* d_out, int out_size, void* d_ws, size_t ws_size,
                              hipStream_t stream) {
  const float* x  = (const float*)d_in[0];
  const float* fp = (const float*)d_in[1];
  float* out = (float*)d_out;

  const size_t NELEM = (size_t)BATCH * SEQ * DMODEL;   // 4,194,304
  bf16_t* xbf = (bf16_t*)d_ws;
  bf16_t* Wbf = xbf + NELEM;                           // BATCH*4*WSZ == NELEM
  bf16_t* Qb  = Wbf + NELEM;
  bf16_t* Kb  = Qb + NELEM;
  bf16_t* Vt  = Kb + NELEM;
  bf16_t* Ct  = Vt + NELEM;

  dim3 blk(256);
  convert_kernel<<<(int)(NELEM / 8 / 256), blk, 0, stream>>>(x, fp, xbf, Wbf);
  qkv_kernel<<<BATCH * (SEQ / 16), blk, 0, stream>>>(xbf, Wbf, fp, Qb, Kb, Vt);
  attn_kernel<<<BATCH * NHEADS * (SEQ / 16) / 8, blk, 0, stream>>>(Qb, Kb, Vt, Ct);
  proj_kernel<<<BATCH * (SEQ / 16), blk, 0, stream>>>(Ct, Wbf, fp, out);
}
